// Model_55886114455856
// MI455X (gfx1250) — matrix operand order repaired and checked by reading, not compiled, not run on hardware
//
#include <hip/hip_runtime.h>

// ---------------------------------------------------------------------------
// Model dims
// ---------------------------------------------------------------------------
#define HOR    16
#define BATCH  512
#define HB     (HOR * BATCH)      // 8192
#define NL     10
#define NR     11
#define DST    29
#define LSTMD  256
#define NTW    4                  // N tiles per wave in the WMMA GEMM

typedef __attribute__((ext_vector_type(16))) _Float16 v16h;
typedef __attribute__((ext_vector_type(8)))  float    v8f;

// ---------------------------------------------------------------------------
// Small helpers
// ---------------------------------------------------------------------------
__device__ __forceinline__ float dot64(const float* a, const float* b) {
  float s = 0.f;
#pragma unroll
  for (int k = 0; k < 64; ++k) s += a[k] * b[k];
  return s;
}

__device__ __forceinline__ float sigmf(float x) { return 1.f / (1.f + __expf(-x)); }

__device__ __forceinline__ float gumbel_noise(unsigned row, unsigned i, unsigned j, unsigned salt) {
  unsigned x = row * 0x9E3779B9u ^ (i * 0x85EBCA6Bu) ^ (j * 0xC2B2AE35u) ^ (salt * 0x27D4EB2Fu);
  x ^= x >> 16; x *= 0x7FEB352Du; x ^= x >> 15; x *= 0x846CA68Bu; x ^= x >> 16;
  float u = (float)x * 2.3283064e-10f;       // [0,1)
  u = u * 0.99999988f + 1e-7f;               // (0,1)
  return -__logf(-__logf(u));
}

// ---------------------------------------------------------------------------
// f32 -> f16 convert / f32 copy / bias combine
// ---------------------------------------------------------------------------
__global__ void cvt_f16_kernel(const float* __restrict__ in, _Float16* __restrict__ out, int n) {
  for (int i = blockIdx.x * blockDim.x + threadIdx.x; i < n; i += gridDim.x * blockDim.x)
    out[i] = (_Float16)in[i];
}
__global__ void copy_f32_kernel(const float* __restrict__ in, float* __restrict__ out, int n) {
  for (int i = blockIdx.x * blockDim.x + threadIdx.x; i < n; i += gridDim.x * blockDim.x)
    out[i] = in[i];
}
__global__ void addb_kernel(const float* __restrict__ a, const float* __restrict__ b,
                            float* __restrict__ out, int n) {
  for (int i = blockIdx.x * blockDim.x + threadIdx.x; i < n; i += gridDim.x * blockDim.x)
    out[i] = a[i] + b[i];
}

// ---------------------------------------------------------------------------
// WMMA GEMM:  C[M][N] = A[M][K] @ W[N][K]^T (+ bias[N])
// A,W f16 row-major (K contiguous), C f32. K%32==0, M%16==0.
// W must be padded to gridDim.x * 16 * 16 rows (pad contents arbitrary: WMMA
// output columns are independent and pad columns are never stored).
// Each wave owns NTW consecutive 16x16 C tiles, fully unconditional K loop:
// 1 A-fragment load feeds NTW back-to-back v_wmma_f32_16x16x32_f16 ops.
// ---------------------------------------------------------------------------
__global__ __launch_bounds__(128) void wmma_gemm_xWT(
    const _Float16* __restrict__ A, const _Float16* __restrict__ W,
    const float* __restrict__ bias, float* __restrict__ C,
    int M, int N, int K)
{
  const int wave  = threadIdx.x >> 5;
  const int lane  = threadIdx.x & 31;
  const int tile0 = (blockIdx.x * 4 + wave) * NTW;   // first N tile of this wave
  const int tileM = blockIdx.y;
  const int l     = lane & 15;
  const int half  = lane >> 4;

  const _Float16* __restrict__ arowp = A + (size_t)(tileM * 16 + l) * K;
  const _Float16* brow[NTW];
#pragma unroll
  for (int t = 0; t < NTW; ++t)
    brow[t] = W + (size_t)((tile0 + t) * 16 + l) * K;

  v8f acc[NTW];
#pragma unroll
  for (int t = 0; t < NTW; ++t) acc[t] = (v8f){};

  for (int k0 = 0; k0 < K; k0 += 32) {
    if (k0 + 32 < K) {
      __builtin_prefetch(arowp + k0 + 32, 0, 3);     // global_prefetch_b8
      __builtin_prefetch(brow[0] + k0 + 32, 0, 3);
    }
    v16h a;
#pragma unroll
    for (int j = 0; j < 16; ++j) {
      a[j]     = arowp[k0 + half * 16 + j];          // K = 0..15 / 16..31
    }
#pragma unroll
    for (int t = 0; t < NTW; ++t) {
      v16h b;
#pragma unroll
      for (int j = 0; j < 16; ++j) b[j] = brow[t][k0 + half * 16 + j];
      acc[t] = __builtin_amdgcn_wmma_f32_16x16x32_f16(
          false, a, false, b, (short)0, acc[t], false, false);
    }
  }

  const int crow0 = tileM * 16 + half * 8;
#pragma unroll
  for (int t = 0; t < NTW; ++t) {
    const int ccol = (tile0 + t) * 16 + l;
    if (ccol < N) {                                  // epilogue-only guard
      const float bv = bias ? bias[ccol] : 0.f;
#pragma unroll
      for (int r = 0; r < 8; ++r)
        C[(size_t)(crow0 + r) * N + ccol] = acc[t][r] + bv;
    }
  }
}

// ---------------------------------------------------------------------------
// Situation embeddings: linear(K->64) + LayerNorm(64) + ReLU, f16 into emb slot.
// One block (64 threads) per (H*B) row; does match/psit/bsit (slots 0/1/2).
// ---------------------------------------------------------------------------
__device__ __forceinline__ void embed_group(
    const float* __restrict__ x, int K,
    const float* __restrict__ W, const float* __restrict__ bb,
    const float* __restrict__ g, const float* __restrict__ be,
    _Float16* __restrict__ dst, float* sv, float* sr, int c)
{
  float acc = bb[c];
  for (int k = 0; k < K; ++k) acc += x[k] * W[c * K + k];
  sv[c] = acc; sr[c] = acc; __syncthreads();
  for (int o = 32; o > 0; o >>= 1) { if (c < o) sr[c] += sr[c + o]; __syncthreads(); }
  const float mu = sr[0] * (1.f / 64.f); __syncthreads();
  const float d = sv[c] - mu; sr[c] = d * d; __syncthreads();
  for (int o = 32; o > 0; o >>= 1) { if (c < o) sr[c] += sr[c + o]; __syncthreads(); }
  const float inv = rsqrtf(sr[0] * (1.f / 64.f) + 1e-5f); __syncthreads();
  float v = d * inv * g[c] + be[c];
  dst[c] = (_Float16)(v > 0.f ? v : 0.f);
}

__global__ __launch_bounds__(64) void embed_kernel(
    const float* __restrict__ msit, const float* __restrict__ psit, const float* __restrict__ bsit,
    const float* __restrict__ WmW, const float* __restrict__ Wmb,
    const float* __restrict__ lnmg, const float* __restrict__ lnmb,
    const float* __restrict__ WpW, const float* __restrict__ Wpb,
    const float* __restrict__ lnpg, const float* __restrict__ lnpb,
    const float* __restrict__ WbW, const float* __restrict__ Wbb,
    const float* __restrict__ lnbg, const float* __restrict__ lnbb,
    _Float16* __restrict__ emb16)
{
  const int row = blockIdx.x;
  const int c = threadIdx.x;
  __shared__ float sv[64], sr[64];
  _Float16* e = emb16 + (size_t)row * 576;
  embed_group(msit + (size_t)row * 10, 10, WmW, Wmb, lnmg, lnmb, e + 0 * 64, sv, sr, c);
  embed_group(psit + (size_t)row * 19, 19, WpW, Wpb, lnpg, lnpb, e + 1 * 64, sv, sr, c);
  embed_group(bsit + (size_t)row * 15, 15, WbW, Wbb, lnbg, lnbb, e + 2 * 64, sv, sr, c);
}

// ---------------------------------------------------------------------------
// Fused attention (attack + defence mirror), one wave32 per (H*B) row.
// All Q/K/V projections, softmaxes, gumbel-hard argmax kept in LDS (~46 KB).
// Writes emb slots 3..8 as f16.
// ---------------------------------------------------------------------------
__global__ __launch_bounds__(32) void attn_kernel(
    const float* __restrict__ ps_g, const float* __restrict__ bs_g,
    const float* __restrict__ lts_g, const float* __restrict__ rts_g,
    const float* __restrict__ q1aW, const float* __restrict__ q1ab,
    const float* __restrict__ k1aW, const float* __restrict__ k1ab,
    const float* __restrict__ v1aW, const float* __restrict__ v1ab,
    const float* __restrict__ q2aW, const float* __restrict__ q2ab,
    const float* __restrict__ k2aW, const float* __restrict__ k2ab,
    const float* __restrict__ q1dW, const float* __restrict__ q1db,
    const float* __restrict__ k1dW, const float* __restrict__ k1db,
    const float* __restrict__ v1dW, const float* __restrict__ v1db,
    const float* __restrict__ q2dW, const float* __restrict__ q2db,
    const float* __restrict__ k2dW, const float* __restrict__ k2db,
    _Float16* __restrict__ emb16)
{
  const int b = blockIdx.x;
  const int lane = threadIdx.x;

  __shared__ float s_ps[DST], s_bs[DST], s_lts[NL][DST], s_rts[NR][DST];
  __shared__ float s_pq[64], s_pk[64], s_pv[64];
  __shared__ float s_bq[64], s_bk[64], s_bv[64];
  __shared__ float s_ltq[NL][64], s_ltka[NL][64], s_ltva[NL][64], s_ltkd[NL][64], s_ltvd[NL][64];
  __shared__ float s_rtq[NR][64], s_rtka[NR][64], s_rtva[NR][64], s_rtkd[NR][64], s_rtvd[NR][64];
  __shared__ float s_prem[64], s_blem[64];
  __shared__ float s_ltre[NL][64], s_rtle[NR][64];
  __shared__ float s_q2[64], s_ltk2[NL][64], s_bq2[64], s_rtk2[NR][64];
  __shared__ float s_logA[16], s_logB[16];
  __shared__ float s_ltr[NL][NR], s_rtl[NR][NL];
  __shared__ float s_lra1[NL][NR], s_rla1[NR][NL];
  __shared__ float s_selfA[NL], s_selfD[NR];
  __shared__ float s_a1rA[NL][NR + 1], s_a1lD[NR][NL + 1];
  __shared__ int   s_hidxA[NL], s_hidxD[NR];
  __shared__ float s_pl2[NL], s_br2[NR];

  // ---- stage states ----
  for (int i = lane; i < DST; i += 32) { s_ps[i] = ps_g[(size_t)b * DST + i]; s_bs[i] = bs_g[(size_t)b * DST + i]; }
  for (int i = lane; i < NL * DST; i += 32) ((float*)s_lts)[i] = lts_g[(size_t)b * NL * DST + i];
  for (int i = lane; i < NR * DST; i += 32) ((float*)s_rts)[i] = rts_g[(size_t)b * NR * DST + i];
  __syncthreads();

  // ---- projections 29 -> 64 (each lane owns cols lane, lane+32) ----
  for (int cc = 0; cc < 2; ++cc) {
    const int c = lane + cc * 32;
    {
      float aq = q1ab[c], ak = k1ab[c], av = v1ab[c];
#pragma unroll
      for (int k = 0; k < DST; ++k) {
        const float x = s_ps[k];
        aq += x * q1aW[c * DST + k]; ak += x * k1aW[c * DST + k]; av += x * v1aW[c * DST + k];
      }
      s_pq[c] = aq; s_pk[c] = ak; s_pv[c] = av;
    }
    {
      float aq = q1db[c], ak = k1db[c], av = v1db[c];
#pragma unroll
      for (int k = 0; k < DST; ++k) {
        const float x = s_bs[k];
        aq += x * q1dW[c * DST + k]; ak += x * k1dW[c * DST + k]; av += x * v1dW[c * DST + k];
      }
      s_bq[c] = aq; s_bk[c] = ak; s_bv[c] = av;
    }
    for (int n = 0; n < NL; ++n) {
      float aq = q1ab[c], aka = k1ab[c], ava = v1ab[c], akd = k1db[c], avd = v1db[c];
#pragma unroll
      for (int k = 0; k < DST; ++k) {
        const float x = s_lts[n][k];
        aq += x * q1aW[c * DST + k]; aka += x * k1aW[c * DST + k]; ava += x * v1aW[c * DST + k];
        akd += x * k1dW[c * DST + k]; avd += x * v1dW[c * DST + k];
      }
      s_ltq[n][c] = aq; s_ltka[n][c] = aka; s_ltva[n][c] = ava; s_ltkd[n][c] = akd; s_ltvd[n][c] = avd;
    }
    for (int n = 0; n < NR; ++n) {
      float aq = q1db[c], aka = k1ab[c], ava = v1ab[c], akd = k1db[c], avd = v1db[c];
#pragma unroll
      for (int k = 0; k < DST; ++k) {
        const float x = s_rts[n][k];
        aq += x * q1dW[c * DST + k]; aka += x * k1aW[c * DST + k]; ava += x * v1aW[c * DST + k];
        akd += x * k1dW[c * DST + k]; avd += x * v1dW[c * DST + k];
      }
      s_rtq[n][c] = aq; s_rtka[n][c] = aka; s_rtva[n][c] = ava; s_rtkd[n][c] = akd; s_rtvd[n][c] = avd;
    }
  }
  __syncthreads();

  // ---- first-hop single-query attentions ----
  if (lane < 1 + NR) s_logA[lane] = dot64(s_pq, lane == 0 ? s_pk : s_rtka[lane - 1]) * 0.125f;
  if (lane < 1 + NL) s_logB[lane] = dot64(s_bq, lane == 0 ? s_bk : s_ltkd[lane - 1]) * 0.125f;
  __syncthreads();

  float attA[1 + NR], attB[1 + NL];
  {
    float m = -1e30f;
#pragma unroll
    for (int j = 0; j < 1 + NR; ++j) m = fmaxf(m, s_logA[j]);
    float s = 0.f;
#pragma unroll
    for (int j = 0; j < 1 + NR; ++j) { attA[j] = __expf(s_logA[j] - m); s += attA[j]; }
    const float r = 1.f / s;
#pragma unroll
    for (int j = 0; j < 1 + NR; ++j) attA[j] *= r;
  }
  {
    float m = -1e30f;
#pragma unroll
    for (int j = 0; j < 1 + NL; ++j) m = fmaxf(m, s_logB[j]);
    float s = 0.f;
#pragma unroll
    for (int j = 0; j < 1 + NL; ++j) { attB[j] = __expf(s_logB[j] - m); s += attB[j]; }
    const float r = 1.f / s;
#pragma unroll
    for (int j = 0; j < 1 + NL; ++j) attB[j] *= r;
  }
  for (int cc = 0; cc < 2; ++cc) {
    const int c = lane + cc * 32;
    float v = s_pv[c] + attA[0] * s_pv[c];
#pragma unroll
    for (int j = 0; j < NR; ++j) v += attA[1 + j] * s_rtva[j][c];
    s_prem[c] = v;
    emb16[(size_t)b * 576 + 3 * 64 + c] = (_Float16)v;

    float w = s_bv[c] + attB[0] * s_bv[c];
#pragma unroll
    for (int j = 0; j < NL; ++j) w += attB[1 + j] * s_ltvd[j][c];
    s_blem[c] = w;
    emb16[(size_t)b * 576 + 6 * 64 + c] = (_Float16)w;
  }

  // ---- cross logits + self logits ----
  for (int e = lane; e < NL * NR; e += 32) {
    const int n = e / NR, k = e % NR;
    s_ltr[n][k] = dot64(s_ltq[n], s_rtka[k]) * 0.125f;
  }
  for (int e = lane; e < NR * NL; e += 32) {
    const int n = e / NL, k = e % NL;
    s_rtl[n][k] = dot64(s_rtq[n], s_ltkd[k]) * 0.125f;
  }
  if (lane < NL) s_selfA[lane] = dot64(s_ltq[lane], s_ltka[lane]) * 0.125f;
  if (lane < NR) s_selfD[lane] = dot64(s_rtq[lane], s_rtkd[lane]) * 0.125f;
  __syncthreads();

  // ---- per-row softmaxes + gumbel-hard argmax (forward == one_hot(argmax)) ----
  if (lane < NL) {
    const int n = lane;
    float m = -1e30f;
#pragma unroll
    for (int k = 0; k < NR; ++k) m = fmaxf(m, s_ltr[n][k]);
    float ex[NR], s = 0.f;
#pragma unroll
    for (int k = 0; k < NR; ++k) { ex[k] = __expf(s_ltr[n][k] - m); s += ex[k]; }
    const float r = 1.f / s;
#pragma unroll
    for (int k = 0; k < NR; ++k) s_lra1[n][k] = ex[k] * r;
    int best = 0; float bv = -1e30f;
#pragma unroll
    for (int k = 0; k < NR; ++k) {
      const float t = s_ltr[n][k] + gumbel_noise((unsigned)b, (unsigned)n, (unsigned)k, 1u);
      if (t > bv) { bv = t; best = k; }
    }
    s_hidxA[n] = best;
    const float m2 = fmaxf(m, s_selfA[n]);
    const float e0 = __expf(s_selfA[n] - m2);
    float s2 = e0, ex2[NR];
#pragma unroll
    for (int k = 0; k < NR; ++k) { ex2[k] = __expf(s_ltr[n][k] - m2); s2 += ex2[k]; }
    const float r2 = 1.f / s2;
    s_a1rA[n][0] = e0 * r2;
#pragma unroll
    for (int k = 0; k < NR; ++k) s_a1rA[n][1 + k] = ex2[k] * r2;
  }
  if (lane < NR) {
    const int n = lane;
    float m = -1e30f;
#pragma unroll
    for (int k = 0; k < NL; ++k) m = fmaxf(m, s_rtl[n][k]);
    float ex[NL], s = 0.f;
#pragma unroll
    for (int k = 0; k < NL; ++k) { ex[k] = __expf(s_rtl[n][k] - m); s += ex[k]; }
    const float r = 1.f / s;
#pragma unroll
    for (int k = 0; k < NL; ++k) s_rla1[n][k] = ex[k] * r;
    int best = 0; float bv = -1e30f;
#pragma unroll
    for (int k = 0; k < NL; ++k) {
      const float t = s_rtl[n][k] + gumbel_noise((unsigned)b, (unsigned)n, (unsigned)k, 3u);
      if (t > bv) { bv = t; best = k; }
    }
    s_hidxD[n] = best;
    const float m2 = fmaxf(m, s_selfD[n]);
    const float e0 = __expf(s_selfD[n] - m2);
    float s2 = e0, ex2[NL];
#pragma unroll
    for (int k = 0; k < NL; ++k) { ex2[k] = __expf(s_rtl[n][k] - m2); s2 += ex2[k]; }
    const float r2 = 1.f / s2;
    s_a1lD[n][0] = e0 * r2;
#pragma unroll
    for (int k = 0; k < NL; ++k) s_a1lD[n][1 + k] = ex2[k] * r2;
  }
  __syncthreads();

  // ---- first-hop team embeds ----
  for (int cc = 0; cc < 2; ++cc) {
    const int c = lane + cc * 32;
    for (int n = 0; n < NL; ++n) {
      float v = s_a1rA[n][0] * s_ltva[n][c] + s_ltva[n][c];
#pragma unroll
      for (int k = 0; k < NR; ++k) v += s_a1rA[n][1 + k] * s_rtva[k][c];
      s_ltre[n][c] = v;
    }
    for (int n = 0; n < NR; ++n) {
      float v = s_a1lD[n][0] * s_rtvd[n][c] + s_rtvd[n][c];
#pragma unroll
      for (int k = 0; k < NL; ++k) v += s_a1lD[n][1 + k] * s_ltvd[k][c];
      s_rtle[n][c] = v;
    }
  }
  __syncthreads();

  // ---- second-hop projections 64 -> 64 ----
  for (int cc = 0; cc < 2; ++cc) {
    const int c = lane + cc * 32;
    s_q2[c]  = q2ab[c] + dot64(s_prem, q2aW + c * 64);
    s_bq2[c] = q2db[c] + dot64(s_blem, q2dW + c * 64);
    for (int n = 0; n < NL; ++n) s_ltk2[n][c] = k2ab[c] + dot64(s_ltre[n], k2aW + c * 64);
    for (int n = 0; n < NR; ++n) s_rtk2[n][c] = k2db[c] + dot64(s_rtle[n], k2dW + c * 64);
  }
  __syncthreads();
  if (lane < NL) s_pl2[lane] = dot64(s_q2, s_ltk2[lane]) * 0.125f;
  if (lane < NR) s_br2[lane] = dot64(s_bq2, s_rtk2[lane]) * 0.125f;
  __syncthreads();

  // ---- second-hop softmax + hard, final outputs ----
  float p2A[NL]; int idx2A;
  {
    float m = -1e30f;
#pragma unroll
    for (int n = 0; n < NL; ++n) m = fmaxf(m, s_pl2[n]);
    float s = 0.f;
#pragma unroll
    for (int n = 0; n < NL; ++n) { p2A[n] = __expf(s_pl2[n] - m); s += p2A[n]; }
    const float r = 1.f / s;
#pragma unroll
    for (int n = 0; n < NL; ++n) p2A[n] *= r;
    int best = 0; float bv = -1e30f;
#pragma unroll
    for (int n = 0; n < NL; ++n) {
      const float t = s_pl2[n] + gumbel_noise((unsigned)b, 0u, (unsigned)n, 2u);
      if (t > bv) { bv = t; best = n; }
    }
    idx2A = best;
  }
  float p2D[NR]; int idx2D;
  {
    float m = -1e30f;
#pragma unroll
    for (int n = 0; n < NR; ++n) m = fmaxf(m, s_br2[n]);
    float s = 0.f;
#pragma unroll
    for (int n = 0; n < NR; ++n) { p2D[n] = __expf(s_br2[n] - m); s += p2D[n]; }
    const float r = 1.f / s;
#pragma unroll
    for (int n = 0; n < NR; ++n) p2D[n] *= r;
    int best = 0; float bv = -1e30f;
#pragma unroll
    for (int n = 0; n < NR; ++n) {
      const float t = s_br2[n] + gumbel_noise((unsigned)b, 0u, (unsigned)n, 4u);
      if (t > bv) { bv = t; best = n; }
    }
    idx2D = best;
  }
  float lra2[NR];
#pragma unroll
  for (int k = 0; k < NR; ++k) {
    float s = 0.f;
#pragma unroll
    for (int n = 0; n < NL; ++n) s += p2A[n] * s_lra1[n][k];
    lra2[k] = s;
  }
  float rla2[NL];
#pragma unroll
  for (int k = 0; k < NL; ++k) {
    float s = 0.f;
#pragma unroll
    for (int n = 0; n < NR; ++n) s += p2D[n] * s_rla1[n][k];
    rla2[k] = s;
  }
  const int hkA = s_hidxA[idx2A];
  const int hkD = s_hidxD[idx2D];

  for (int cc = 0; cc < 2; ++cc) {
    const int c = lane + cc * 32;
    float ltp = s_ltva[idx2A][c];
#pragma unroll
    for (int n = 0; n < NL; ++n) ltp += p2A[n] * s_ltva[n][c];
    emb16[(size_t)b * 576 + 4 * 64 + c] = (_Float16)ltp;

    float rtl = s_rtva[hkA][c];
#pragma unroll
    for (int k = 0; k < NR; ++k) rtl += lra2[k] * s_rtva[k][c];
    emb16[(size_t)b * 576 + 5 * 64 + c] = (_Float16)rtl;

    float rtb = s_rtvd[idx2D][c];
#pragma unroll
    for (int n = 0; n < NR; ++n) rtb += p2D[n] * s_rtvd[n][c];
    emb16[(size_t)b * 576 + 7 * 64 + c] = (_Float16)rtb;

    float ltr = s_ltvd[hkD][c];
#pragma unroll
    for (int k = 0; k < NL; ++k) ltr += rla2[k] * s_ltvd[k][c];
    emb16[(size_t)b * 576 + 8 * 64 + c] = (_Float16)ltr;
  }
}

// ---------------------------------------------------------------------------
// LayerNorm(dim) + ReLU. Optional f32 (may alias input) and/or f16 outputs.
// ---------------------------------------------------------------------------
__global__ __launch_bounds__(128) void ln_relu_kernel(
    const float* __restrict__ in, const float* __restrict__ g, const float* __restrict__ b,
    int dim, float* __restrict__ outf, _Float16* __restrict__ outh)
{
  const int row = blockIdx.x;
  const float* x = in + (size_t)row * dim;
  __shared__ float red[128];
  float s = 0.f;
  for (int i = threadIdx.x; i < dim; i += 128) s += x[i];
  red[threadIdx.x] = s; __syncthreads();
  for (int o = 64; o > 0; o >>= 1) { if (threadIdx.x < o) red[threadIdx.x] += red[threadIdx.x + o]; __syncthreads(); }
  const float mu = red[0] / (float)dim; __syncthreads();
  float s2 = 0.f;
  for (int i = threadIdx.x; i < dim; i += 128) { const float d = x[i] - mu; s2 += d * d; }
  red[threadIdx.x] = s2; __syncthreads();
  for (int o = 64; o > 0; o >>= 1) { if (threadIdx.x < o) red[threadIdx.x] += red[threadIdx.x + o]; __syncthreads(); }
  const float inv = rsqrtf(red[0] / (float)dim + 1e-5f); __syncthreads();
  for (int i = threadIdx.x; i < dim; i += 128) {
    float v = (x[i] - mu) * inv * g[i] + b[i];
    v = v > 0.f ? v : 0.f;
    if (outf) outf[(size_t)row * dim + i] = v;
    if (outh) outh[(size_t)row * dim + i] = (_Float16)v;
  }
}

// ---------------------------------------------------------------------------
// LSTM gate fuse (torch gate order i,f,g,o). h written directly as f16.
// ---------------------------------------------------------------------------
__global__ void lstm_gate_kernel(const float* __restrict__ X, const float* __restrict__ Z,
                                 float* __restrict__ c, _Float16* __restrict__ h16)
{
  const int idx = blockIdx.x * blockDim.x + threadIdx.x;
  if (idx >= BATCH * LSTMD) return;
  const int r = idx >> 8, j = idx & 255;
  const float* zx = X + (size_t)r * 1024;
  const float* zz = Z + (size_t)r * 1024;
  const float zi = zx[j]       + zz[j];
  const float zf = zx[256 + j] + zz[256 + j];
  const float zg = zx[512 + j] + zz[512 + j];
  const float zo = zx[768 + j] + zz[768 + j];
  const float cn = sigmf(zf) * c[idx] + sigmf(zi) * tanhf(zg);
  c[idx] = cn;
  h16[idx] = (_Float16)(sigmf(zo) * tanhf(cn));
}

// ---------------------------------------------------------------------------
// Heads
// ---------------------------------------------------------------------------
__global__ __launch_bounds__(32) void head_a_kernel(
    const float* __restrict__ a1r, const float* __restrict__ W, const float* __restrict__ bb,
    const float* __restrict__ avail, float* __restrict__ prob)
{
  const int row = blockIdx.x, lane = threadIdx.x;
  __shared__ float sl[12];
  const float* x = a1r + (size_t)row * 164;
  if (lane < 12) {
    float acc = bb[lane];
    const float* w = W + lane * 164;
    for (int k = 0; k < 164; ++k) acc += x[k] * w[k];
    acc += (avail[(size_t)row * 12 + lane] - 1.f) * 1e7f;
    sl[lane] = acc;
  }
  __syncthreads();
  if (lane < 12) {
    float m = sl[0];
#pragma unroll
    for (int j = 1; j < 12; ++j) m = fmaxf(m, sl[j]);
    float s = 0.f;
#pragma unroll
    for (int j = 0; j < 12; ++j) s += __expf(sl[j] - m);
    prob[(size_t)row * 12 + lane] = __expf(sl[lane] - m) / s;
  }
}

__global__ __launch_bounds__(32) void head_m_kernel(
    const float* __restrict__ m1r, const float* __restrict__ W, const float* __restrict__ bb,
    float* __restrict__ prob)
{
  const int row = blockIdx.x, lane = threadIdx.x;
  __shared__ float sl[8];
  const float* x = m1r + (size_t)row * 164;
  if (lane < 8) {
    float acc = bb[lane];
    const float* w = W + lane * 164;
    for (int k = 0; k < 164; ++k) acc += x[k] * w[k];
    sl[lane] = acc;
  }
  __syncthreads();
  if (lane < 8) {
    float m = sl[0];
#pragma unroll
    for (int j = 1; j < 8; ++j) m = fmaxf(m, sl[j]);
    float s = 0.f;
#pragma unroll
    for (int j = 0; j < 8; ++j) s += __expf(sl[j] - m);
    prob[(size_t)row * 8 + lane] = __expf(sl[lane] - m) / s;
  }
}

__global__ void head_v_kernel(const float* __restrict__ v1r, const float* __restrict__ W,
                              float* __restrict__ out, int rows)
{
  const int r = blockIdx.x * blockDim.x + threadIdx.x;
  if (r >= rows) return;
  const float* x = v1r + (size_t)r * 164;
  float s = 0.f;
  for (int k = 0; k < 164; ++k) s += x[k] * W[k];
  out[r] = s;
}

// ---------------------------------------------------------------------------
// Host driver
// ---------------------------------------------------------------------------
static inline void launch_gemm(const _Float16* A, const _Float16* W, const float* bias,
                               float* C, int M, int N, int K, hipStream_t s)
{
  const int tilesN = (N + 15) / 16;
  const int tilesPerBlock = 4 * NTW;                 // 4 waves x NTW tiles = 16
  dim3 grid((tilesN + tilesPerBlock - 1) / tilesPerBlock, M / 16);
  wmma_gemm_xWT<<<grid, 128, 0, s>>>(A, W, bias, C, M, N, K);
}

// params (d_in[10 + i]) in setup_inputs() leaf order
enum {
  P_FC_MATCH_W = 0, P_FC_MATCH_B, P_LN_MATCH_G, P_LN_MATCH_B,
  P_FC_PSIT_W, P_FC_PSIT_B, P_LN_PSIT_G, P_LN_PSIT_B,
  P_FC_BSIT_W, P_FC_BSIT_B, P_LN_BSIT_G, P_LN_BSIT_B,
  P_Q1A_W, P_Q1A_B, P_K1A_W, P_K1A_B, P_V1A_W, P_V1A_B,
  P_Q2A_W, P_Q2A_B, P_K2A_W, P_K2A_B,
  P_Q1D_W, P_Q1D_B, P_K1D_W, P_K1D_B, P_V1D_W, P_V1D_B,
  P_Q2D_W, P_Q2D_B, P_K2D_W, P_K2D_B,
  P_FC_CAT_W, P_FC_CAT_B, P_LN_CAT_G, P_LN_CAT_B,
  P_W_IH, P_W_HH, P_B_IH, P_B_HH,
  P_PI_A1_W, P_PI_A1_B, P_LN_A1_G, P_LN_A1_B, P_PI_A2_W, P_PI_A2_B,
  P_PI_M1_W, P_PI_M1_B, P_LN_M1_G, P_LN_M1_B, P_PI_M2_W, P_PI_M2_B,
  P_V1_W, P_V1_B, P_LN_V1_G, P_LN_V1_B, P_V2_W
};

extern "C" void kernel_launch(void* const* d_in, const int* in_sizes, int n_in,
                              void* d_out, int out_size, void* d_ws, size_t ws_size,
                              hipStream_t stream)
{
  (void)in_sizes; (void)n_in; (void)out_size; (void)ws_size;
  const float* psit  = (const float*)d_in[0];
  const float* bsit  = (const float*)d_in[1];
  const float* msit  = (const float*)d_in[2];
  const float* pst   = (const float*)d_in[3];
  const float* bst   = (const float*)d_in[4];
  const float* ltst  = (const float*)d_in[5];
  const float* rtst  = (const float*)d_in[6];
  const float* avail = (const float*)d_in[7];
  const float* h_in  = (const float*)d_in[8];
  const float* c_in  = (const float*)d_in[9];
#define PRM(i) ((const float*)d_in[10 + (i)])

  // ---- workspace bump allocator ----
  char* w = (char*)d_ws;
  auto alloc = [&](size_t bytes) -> void* {
    void* p = (void*)w;
    w += (bytes + 255) & ~(size_t)255;
    return p;
  };
  _Float16* fcW16  = (_Float16*)alloc((size_t)256 * 576 * 2);
  _Float16* Wih16  = (_Float16*)alloc((size_t)1024 * 256 * 2);
  _Float16* Whh16  = (_Float16*)alloc((size_t)1024 * 256 * 2);
  _Float16* a1W16  = (_Float16*)alloc((size_t)256 * 256 * 2);  // padded: grid covers 16 tiles
  _Float16* m1W16  = (_Float16*)alloc((size_t)256 * 256 * 2);
  _Float16* v1W16  = (_Float16*)alloc((size_t)256 * 256 * 2);
  _Float16* h16ini = (_Float16*)alloc((size_t)BATCH * LSTMD * 2);
  _Float16* emb16  = (_Float16*)alloc((size_t)HB * 576 * 2);
  _Float16* cat16  = (_Float16*)alloc((size_t)HB * 256 * 2);
  _Float16* out16  = (_Float16*)alloc((size_t)HB * 256 * 2);
  float* catraw = (float*)alloc((size_t)HB * 256 * 4);
  float* X      = (float*)alloc((size_t)HB * 1024 * 4);
  float* Z      = (float*)alloc((size_t)BATCH * 1024 * 4);
  float* cbuf   = (float*)alloc((size_t)BATCH * LSTMD * 4);
  float* biasIH = (float*)alloc((size_t)1024 * 4);
  float* araw   = (float*)alloc((size_t)HB * 164 * 4);
  float* mraw   = (float*)alloc((size_t)HB * 164 * 4);
  float* vraw   = (float*)alloc((size_t)HB * 164 * 4);

  float* prob_out  = (float*)d_out;              // [HB][12]
  float* probm_out = prob_out + (size_t)HB * 12; // [HB][8]
  float* v_out     = prob_out + (size_t)HB * 20; // [HB][1]

  // ---- weight / state conversions ----
  auto cvt = [&](const float* src, _Float16* dst, int n) {
    cvt_f16_kernel<<<(n + 255) / 256, 256, 0, stream>>>(src, dst, n);
  };
  cvt(PRM(P_FC_CAT_W), fcW16, 256 * 576);
  cvt(PRM(P_W_IH),     Wih16, 1024 * 256);
  cvt(PRM(P_W_HH),     Whh16, 1024 * 256);
  cvt(PRM(P_PI_A1_W),  a1W16, 164 * 256);
  cvt(PRM(P_PI_M1_W),  m1W16, 164 * 256);
  cvt(PRM(P_V1_W),     v1W16, 164 * 256);
  cvt(h_in, h16ini, BATCH * LSTMD);
  copy_f32_kernel<<<(BATCH * LSTMD + 255) / 256, 256, 0, stream>>>(c_in, cbuf, BATCH * LSTMD);
  addb_kernel<<<4, 256, 0, stream>>>(PRM(P_B_IH), PRM(P_B_HH), biasIH, 1024);

  // ---- embeddings (emb slots 0..2) ----
  embed_kernel<<<HB, 64, 0, stream>>>(
      msit, psit, bsit,
      PRM(P_FC_MATCH_W), PRM(P_FC_MATCH_B), PRM(P_LN_MATCH_G), PRM(P_LN_MATCH_B),
      PRM(P_FC_PSIT_W),  PRM(P_FC_PSIT_B),  PRM(P_LN_PSIT_G),  PRM(P_LN_PSIT_B),
      PRM(P_FC_BSIT_W),  PRM(P_FC_BSIT_B),  PRM(P_LN_BSIT_G),  PRM(P_LN_BSIT_B),
      emb16);

  // ---- fused attention (emb slots 3..8) ----
  attn_kernel<<<HB, 32, 0, stream>>>(
      pst, bst, ltst, rtst,
      PRM(P_Q1A_W), PRM(P_Q1A_B), PRM(P_K1A_W), PRM(P_K1A_B), PRM(P_V1A_W), PRM(P_V1A_B),
      PRM(P_Q2A_W), PRM(P_Q2A_B), PRM(P_K2A_W), PRM(P_K2A_B),
      PRM(P_Q1D_W), PRM(P_Q1D_B), PRM(P_K1D_W), PRM(P_K1D_B), PRM(P_V1D_W), PRM(P_V1D_B),
      PRM(P_Q2D_W), PRM(P_Q2D_B), PRM(P_K2D_W), PRM(P_K2D_B),
      emb16);

  // ---- fc_cat (WMMA) + LN + ReLU ----
  launch_gemm(emb16, fcW16, PRM(P_FC_CAT_B), catraw, HB, 256, 576, stream);
  ln_relu_kernel<<<HB, 128, 0, stream>>>(catraw, PRM(P_LN_CAT_G), PRM(P_LN_CAT_B), 256, nullptr, cat16);

  // ---- LSTM input projection for all timesteps (WMMA) ----
  launch_gemm(cat16, Wih16, biasIH, X, HB, 1024, 256, stream);

  // ---- 16 recurrent steps: h@W_hh (WMMA) + fused gates ----
  for (int t = 0; t < HOR; ++t) {
    const _Float16* hprev = (t == 0) ? h16ini : (out16 + (size_t)(t - 1) * BATCH * LSTMD);
    launch_gemm(hprev, Whh16, nullptr, Z, BATCH, 1024, 256, stream);
    lstm_gate_kernel<<<(BATCH * LSTMD + 255) / 256, 256, 0, stream>>>(
        X + (size_t)t * BATCH * 1024, Z, cbuf, out16 + (size_t)t * BATCH * LSTMD);
  }

  // ---- heads (WMMA 256->164 padded to 256, LN in-place, tiny final layers) ----
  launch_gemm(out16, a1W16, PRM(P_PI_A1_B), araw, HB, 164, 256, stream);
  ln_relu_kernel<<<HB, 128, 0, stream>>>(araw, PRM(P_LN_A1_G), PRM(P_LN_A1_B), 164, araw, nullptr);
  head_a_kernel<<<HB, 32, 0, stream>>>(araw, PRM(P_PI_A2_W), PRM(P_PI_A2_B), avail, prob_out);

  launch_gemm(out16, m1W16, PRM(P_PI_M1_B), mraw, HB, 164, 256, stream);
  ln_relu_kernel<<<HB, 128, 0, stream>>>(mraw, PRM(P_LN_M1_G), PRM(P_LN_M1_B), 164, mraw, nullptr);
  head_m_kernel<<<HB, 32, 0, stream>>>(mraw, PRM(P_PI_M2_W), PRM(P_PI_M2_B), probm_out);

  launch_gemm(out16, v1W16, PRM(P_V1_B), vraw, HB, 164, 256, stream);
  ln_relu_kernel<<<HB, 128, 0, stream>>>(vraw, PRM(P_LN_V1_G), PRM(P_LN_V1_B), 164, vraw, nullptr);
  head_v_kernel<<<(HB + 255) / 256, 256, 0, stream>>>(vraw, PRM(P_V2_W), v_out, HB);
#undef PRM
}
